// PointNetPlusSetAbstaction_63874753626491
// MI455X (gfx1250) — matrix operand order repaired and checked by reading, not compiled, not run on hardware
//
#include <hip/hip_runtime.h>
#include <hip/hip_bf16.h>
#include <stdint.h>

// ---------------- problem constants ----------------
#define BATCH   16
#define NPTS    4096
#define CPTS    64
#define NPOINT  1024
#define NSAMP   32
#define RAD2    0.04f      // 0.2^2
#define KPAD    96         // 67 padded to 3 k-steps of 32
#define ROWS    128        // 4 centroids * 32 samples per block

typedef __attribute__((ext_vector_type(16))) _Float16 v16h;
typedef __attribute__((ext_vector_type(8)))  float    v8f;

// ======================================================================
// Kernel 1: farthest point sampling (one block per batch)
//   dyn LDS: xyz[4096*3] f32 | mind[4096] f32 | redv[256] | redi[256] | s_far
// ======================================================================
__global__ void fps_kernel(const float* __restrict__ xyz, float* __restrict__ out_xyz) {
    extern __shared__ __align__(16) char smem[];
    float* xyzL = (float*)smem;                      // 49152 B
    float* mind = (float*)(smem + 49152);            // 16384 B
    float* redv = (float*)(smem + 49152 + 16384);    // 1024 B
    int*   redi = (int*)  (smem + 49152 + 16384 + 1024); // 1024 B
    int*   sfar = (int*)  (smem + 49152 + 16384 + 2048);

    const int b   = blockIdx.x;
    const int tid = threadIdx.x;
    const float* xb = xyz + (size_t)b * NPTS * 3;

    for (int i = tid; i < NPTS * 3; i += 256) xyzL[i] = xb[i];
    for (int i = tid; i < NPTS;     i += 256) mind[i] = 1e10f;
    if (tid == 0) *sfar = 0;
    __syncthreads();

    for (int it = 0; it < NPOINT; ++it) {
        const int far = *sfar;
        if (tid < 3) out_xyz[((size_t)b * NPOINT + it) * 3 + tid] = xyzL[far * 3 + tid];
        const float cx = xyzL[far * 3 + 0];
        const float cy = xyzL[far * 3 + 1];
        const float cz = xyzL[far * 3 + 2];

        float bv = -1.f; int bi = 0;
        #pragma unroll 4
        for (int kk = 0; kk < NPTS / 256; ++kk) {
            const int i = tid + kk * 256;
            const float dx = xyzL[3 * i + 0] - cx;
            const float dy = xyzL[3 * i + 1] - cy;
            const float dz = xyzL[3 * i + 2] - cz;
            const float d  = dx * dx + dy * dy + dz * dz;
            float m = fminf(mind[i], d);
            mind[i] = m;
            if (m > bv) { bv = m; bi = i; }
        }
        redv[tid] = bv; redi[tid] = bi;
        __syncthreads();
        for (int off = 128; off > 0; off >>= 1) {
            if (tid < off) {
                const float v2 = redv[tid + off];
                const int   i2 = redi[tid + off];
                if (v2 > redv[tid] || (v2 == redv[tid] && i2 < redi[tid])) {
                    redv[tid] = v2; redi[tid] = i2;
                }
            }
            __syncthreads();
        }
        if (tid == 0) *sfar = redi[0];
        __syncthreads();
    }
}

// ======================================================================
// Kernel 2: ball query (one block per batch); first 32 in-radius indices
//           in ascending index order == reference sort-by-key semantics.
// ======================================================================
__global__ void ball_kernel(const float* __restrict__ xyz,
                            const float* __restrict__ new_xyz,
                            int* __restrict__ out_idx) {
    extern __shared__ __align__(16) char smem[];
    float* xyzL = (float*)smem;            // 49152 B
    const int b   = blockIdx.x;
    const int tid = threadIdx.x;
    const float* xb = xyz + (size_t)b * NPTS * 3;
    for (int i = tid; i < NPTS * 3; i += 256) xyzL[i] = xb[i];
    __syncthreads();

    for (int c = 0; c < NPOINT / 256; ++c) {
        const int ci = c * 256 + tid;
        const size_t sb = (size_t)b * NPOINT + ci;
        const float nx = new_xyz[sb * 3 + 0];
        const float ny = new_xyz[sb * 3 + 1];
        const float nz = new_xyz[sb * 3 + 2];
        int* orow = out_idx + sb * NSAMP;
        int cnt = 0, first = NPTS - 1;
        for (int j = 0; j < NPTS; ++j) {
            const float dx = xyzL[3 * j + 0] - nx;
            const float dy = xyzL[3 * j + 1] - ny;
            const float dz = xyzL[3 * j + 2] - nz;
            if (dx * dx + dy * dy + dz * dz <= RAD2) {
                if (cnt == 0) first = j;
                orow[cnt] = j;
                if (++cnt == NSAMP) break;
            }
        }
        for (; cnt < NSAMP; ++cnt) orow[cnt] = first;
    }
}

// ======================================================================
// Kernel 3: fused gather + 3-layer MLP (WMMA f16->f32) + register maxpool.
//   128 rows (4 centroids) per block, 8 waves, dynamic LDS plan:
//     in0  : 128x96  f16  @ 0      (24576)
//     act1 : 128x64  f16  @ 24576  (16384)
//     act2 : 128x64  f16  @ 40960  (16384)
//     w0t  : 64x96   f16  @ 57344  (12288)   Wt[n][k]
//     w1t  : 64x64   f16  @ 69632  ( 8192)
//     w2t  : 128x64  f16  @ 77824  (16384)
//     bias : 256     f32  @ 94208  ( 1024)   -> total 95232 B
// ======================================================================

// Multi-tile WMMA: one B fragment per k-step reused across MT independent
// accumulators -> back-to-back v_wmma with no RAW hazard NOPs, uniform flow.
template<int MT, int KS>
__device__ __forceinline__ void wmma_block(const _Float16* __restrict__ A, int lda,
                                           const _Float16* __restrict__ Bt, int ldb,
                                           int lane, v8f* acc) {
    const int half = lane >> 4;
    const int l    = lane & 15;
    #pragma unroll
    for (int ks = 0; ks < KS; ++ks) {
        union { v16h v; uint4 q[2]; } bf;
        const _Float16* brow = Bt + l * ldb + ks * 32;
        bf.q[0] = *(const uint4*)(brow + half * 8);       // K = ks*32 + half*8 .. +7
        bf.q[1] = *(const uint4*)(brow + 16 + half * 8);  // K = ks*32 + 16 + half*8 .. +7
        #pragma unroll
        for (int mt = 0; mt < MT; ++mt) {
            union { v16h v; uint4 q[2]; } a;
            const _Float16* arow = A + (mt * 16 + l) * lda + ks * 32;
            a.q[0] = *(const uint4*)(arow + half * 8);      // K = ks*32 + half*8 .. +7
            a.q[1] = *(const uint4*)(arow + 16 + half * 8); // K = ks*32 + 16 + half*8 .. +7
            acc[mt] = __builtin_amdgcn_wmma_f32_16x16x32_f16(false, a.v, false, bf.v,
                                                             (short)0, acc[mt], false, false);
        }
    }
}

__device__ __forceinline__ void store_act(_Float16* dst, int ldd, v8f c,
                                          const float* bias, int lane) {
    const int half = lane >> 4;
    const int n    = lane & 15;
    const float bv = bias[n];
    #pragma unroll
    for (int r = 0; r < 8; ++r) {
        float v = c[r] + bv;
        v = v > 0.f ? v : 0.f;
        dst[(r + 8 * half) * ldd + n] = (_Float16)v;
    }
}

__global__ void mlp_kernel(const float* __restrict__ xyz,
                           const float* __restrict__ points,
                           const float* __restrict__ W0, const float* __restrict__ b0,
                           const float* __restrict__ W1, const float* __restrict__ b1,
                           const float* __restrict__ W2, const float* __restrict__ b2,
                           const float* __restrict__ new_xyz,
                           const int*   __restrict__ idxbuf,
                           float* __restrict__ out_pts) {
    extern __shared__ __align__(16) char smem[];
    _Float16* in0  = (_Float16*)(smem + 0);
    _Float16* act1 = (_Float16*)(smem + 24576);
    _Float16* act2 = (_Float16*)(smem + 40960);
    _Float16* w0t  = (_Float16*)(smem + 57344);
    _Float16* w1t  = (_Float16*)(smem + 69632);
    _Float16* w2t  = (_Float16*)(smem + 77824);
    float*    biasL = (float*) (smem + 94208);

    const int tid  = threadIdx.x;
    const int lane = tid & 31;
    const int wid  = tid >> 5;
    const int b    = blockIdx.x >> 8;        // 256 blocks per batch
    const int s0   = (blockIdx.x & 255) * 4; // 4 centroids per block

    // ---- stage weights (transposed, f16, zero-padded K) + biases ----
    for (int i = tid; i < 64 * KPAD; i += 256) {           // w0t[n][k]
        const int n = i / KPAD, k = i % KPAD;
        w0t[i] = (k < 67) ? (_Float16)W0[k * 64 + n] : (_Float16)0.f;
    }
    for (int i = tid; i < 64 * 64; i += 256) {             // w1t[n][k]
        const int n = i >> 6, k = i & 63;
        w1t[i] = (_Float16)W1[k * 64 + n];
    }
    for (int i = tid; i < 128 * 64; i += 256) {            // w2t[n][k]
        const int n = i >> 6, k = i & 63;
        w2t[i] = (_Float16)W2[k * 128 + n];
    }
    for (int i = tid; i < 256; i += 256)
        biasL[i] = (i < 64) ? b0[i] : (i < 128) ? b1[i - 64] : b2[i - 128];

    // ---- gather features: row r = (centroid ci, sample s) ----
    {
        const int r = tid >> 1, part = tid & 1;
        const int ci = r >> 5, s = r & 31;
        const size_t sb = (size_t)b * NPOINT + (s0 + ci);
        const int pidx = idxbuf[sb * NSAMP + s];
        const float* prow = points + ((size_t)b * NPTS + pidx) * CPTS;
        _Float16* frow = in0 + r * KPAD;
        if (part == 0) {
            const float* xr = xyz + ((size_t)b * NPTS + pidx) * 3;
            frow[0] = (_Float16)(xr[0] - new_xyz[sb * 3 + 0]);
            frow[1] = (_Float16)(xr[1] - new_xyz[sb * 3 + 1]);
            frow[2] = (_Float16)(xr[2] - new_xyz[sb * 3 + 2]);
            for (int c = 3; c < 48; ++c) frow[c] = (_Float16)prow[c - 3];
        } else {
            for (int c = 48; c < 67; ++c) frow[c] = (_Float16)prow[c - 3];
            for (int c = 67; c < KPAD; ++c) frow[c] = (_Float16)0.f;
        }
    }
    __syncthreads();

    // Wave tile ownership (uniform, compile-time trip counts):
    //   layers 1-2: nt = wid&3 (16-col tile), mgrp = wid>>2 (4 row-tiles each)
    //   layer  3  : nt = wid, all 8 row-tiles -> pool in registers
    const int ntA = wid & 3;
    const int mg  = wid >> 2;

    // ---- layer 1: 128x96 @ 96x64  (3 k-steps, 4 M-tiles/wave) ----
    {
        v8f acc[4]; 
        #pragma unroll
        for (int i = 0; i < 4; ++i) acc[i] = (v8f){};
        wmma_block<4, 3>(in0 + mg * 64 * KPAD, KPAD, w0t + ntA * 16 * KPAD, KPAD, lane, acc);
        #pragma unroll
        for (int mt = 0; mt < 4; ++mt)
            store_act(act1 + (mg * 4 + mt) * 16 * 64 + ntA * 16, 64, acc[mt],
                      biasL + ntA * 16, lane);
    }
    __syncthreads();

    // ---- layer 2: 128x64 @ 64x64  (2 k-steps, 4 M-tiles/wave) ----
    {
        v8f acc[4];
        #pragma unroll
        for (int i = 0; i < 4; ++i) acc[i] = (v8f){};
        wmma_block<4, 2>(act1 + mg * 64 * 64, 64, w1t + ntA * 16 * 64, 64, lane, acc);
        #pragma unroll
        for (int mt = 0; mt < 4; ++mt)
            store_act(act2 + (mg * 4 + mt) * 16 * 64 + ntA * 16, 64, acc[mt],
                      biasL + 64 + ntA * 16, lane);
    }
    __syncthreads();

    // ---- layer 3: 128x64 @ 64x128 (2 k-steps, 8 M-tiles/wave) + maxpool ----
    {
        v8f acc[8];
        #pragma unroll
        for (int i = 0; i < 8; ++i) acc[i] = (v8f){};
        wmma_block<8, 2>(act2, 64, w2t + wid * 16 * 64, 64, lane, acc);

        const int n = lane & 15;
        const float bv = biasL[128 + wid * 16 + n];
        #pragma unroll
        for (int ci = 0; ci < 4; ++ci) {
            // rows of centroid ci are M-tiles {2ci, 2ci+1}; lane holds 16 of the
            // 32 rows (half split), merge the other half with a shuffle.
            float m = -1e30f;
            #pragma unroll
            for (int j = 0; j < 2; ++j) {
                const v8f& c = acc[2 * ci + j];
                #pragma unroll
                for (int r = 0; r < 8; ++r) {
                    float v = c[r] + bv;
                    v = v > 0.f ? v : 0.f;
                    m = v > m ? v : m;
                }
            }
            const float mo = __shfl_xor(m, 16, 32);
            m = mo > m ? mo : m;
            if (lane < 16)
                out_pts[((size_t)b * NPOINT + (s0 + ci)) * 128 + wid * 16 + n] = m;
        }
    }
}

// ======================================================================
// launch
// ======================================================================
extern "C" void kernel_launch(void* const* d_in, const int* in_sizes, int n_in,
                              void* d_out, int out_size, void* d_ws, size_t ws_size,
                              hipStream_t stream) {
    const float* xyz    = (const float*)d_in[0];
    const float* points = (const float*)d_in[1];
    const float* W0 = (const float*)d_in[2];
    const float* b0 = (const float*)d_in[3];
    const float* W1 = (const float*)d_in[4];
    const float* b1 = (const float*)d_in[5];
    const float* W2 = (const float*)d_in[6];
    const float* b2 = (const float*)d_in[7];

    float* out_xyz = (float*)d_out;                              // B*NPOINT*3
    float* out_pts = out_xyz + (size_t)BATCH * NPOINT * 3;       // B*NPOINT*128
    int*   out_idx = (int*)(out_pts + (size_t)BATCH * NPOINT * 128); // B*NPOINT*32

    // 1) FPS -> new_xyz
    const size_t lds1 = 49152 + 16384 + 1024 + 1024 + 16;
    fps_kernel<<<BATCH, 256, lds1, stream>>>(xyz, out_xyz);

    // 2) ball query -> idx
    ball_kernel<<<BATCH, 256, 49152, stream>>>(xyz, out_xyz, out_idx);

    // 3) gather + MLP(WMMA) + register maxpool -> new_points
    const int nblk = BATCH * (NPOINT / 4);   // 4096 blocks
    mlp_kernel<<<nblk, 256, 95232, stream>>>(xyz, points, W0, b0, W1, b1, W2, b2,
                                             out_xyz, out_idx, out_pts);
}
